// GptOssGroupedExperts_28252294873407
// MI455X (gfx1250) — compile-verified, matrix operand order checked by reading, not run on hardware
//
#include <hip/hip_runtime.h>

// ---------------------------------------------------------------------------
// GPT-OSS grouped-expert MoE FFN for gfx1250 (MI455X), wave32 + WMMA + TDM.
//
//   gate_up = x @ gate_up_proj[e] + bias[e]        (N,H)x(H,2I) -> (N,2I)
//   h       = (clamp(up)+1) * gate*sigmoid(1.702*gate)          -> (N,I)  [f16 ws]
//   out     = h @ down_proj[e] + bias[e]           (N,I)x(I,H)  -> (N,H)
//
// Math path: f32 -> f16 tile conversion (v_cvt_pk_rtz_f16_f32),
// v_wmma_f32_16x16x32_f16 with f32 accumulation.  Compute-bound
// (AI ~680 flop/B vs ~100 flop/B machine balance at 23.3 TB/s); per-expert
// weights (66 MB) stay resident in the 192 MB L2 across the 256 consecutive
// row tiles of each expert.  K-step = 64 (8 WMMA per barrier in k1).
// Kernel 2 stages its f16 A tile with the Tensor Data Mover
// (tensor_load_to_lds + s_wait_tensorcnt), using D# LDS padding to match the
// bank-padded LDS stride.
// ---------------------------------------------------------------------------

typedef __attribute__((ext_vector_type(16))) _Float16 v16h;
typedef __attribute__((ext_vector_type(2)))  __fp16   v2fp16;  // cvt_pkrtz result type
typedef __attribute__((ext_vector_type(8)))  float    v8f;
typedef __attribute__((ext_vector_type(4)))  unsigned int v4u;
typedef __attribute__((ext_vector_type(8)))  int      v8i;
typedef __attribute__((ext_vector_type(4)))  int      v4i;

#define HDIM  2880
#define IDIM  2880
#define NEXP  8
#define ALPHA 1.702f
#define LIMIT 7.0f
#define KSTEP 64
#define LDS_STRIDE 72   // halves per LDS row: 64 K + 8 pad (even -> 4B-aligned pairs)

__device__ __forceinline__ int expert_of_row(const int* __restrict__ counts, int row) {
  int s = 0;
  int e = NEXP - 1;
#pragma unroll
  for (int i = 0; i < NEXP; ++i) {
    s += counts[i];
    if (row < s) { e = i; break; }
  }
  return e;
}

// Single-instruction f32x2 -> packed f16x2 (v_cvt_pk_rtz_f16_f32).
__device__ __forceinline__ unsigned pack2(float lo, float hi) {
  union { v2fp16 h2; unsigned u; } p;
  p.h2 = __builtin_amdgcn_cvt_pkrtz(lo, hi);
  return p.u;
}

// 16x32 f16 fragment load (A layout; B identical with N in place of M since
// LDS stores B column-major [col][K]).  ISA 16-bit A 16x32 layout:
// lane&15 = row; lanes 0-15 carry K 0-7 & 16-23, lanes 16-31 carry K 8-15 &
// 24-31; each VGPR packs 2 consecutive K halves -> 2x ds_load_b128 per frag.
__device__ __forceinline__ v16h load_frag(const _Float16* __restrict__ base, int lane) {
  union { v16h v; unsigned u[8]; } f;
  const int r  = lane & 15;
  const int kb = (lane >> 4) << 3;       // 0 or 8
  const _Float16* row = base + r * LDS_STRIDE;
#pragma unroll
  for (int i = 0; i < 4; ++i) {
    f.u[i]     = *reinterpret_cast<const unsigned*>(row + kb + 2 * i);       // K 0..15
    f.u[i + 4] = *reinterpret_cast<const unsigned*>(row + 16 + kb + 2 * i);  // K 16..31
  }
  return f.v;
}

__device__ __forceinline__ v8f wmma16(v16h a, v16h b, v8f c) {
  // (neg_a, A, neg_b, B, c_mod, C, reuse_a, reuse_b)
  return __builtin_amdgcn_wmma_f32_16x16x32_f16(false, a, false, b, (short)0, c,
                                                false, false);
}

// Issue a TDM 2D tile load: [rows x kHalves] f16 tile from `gsrc` (row stride
// `gstride` halves) into LDS at byte offset `ldsOff`, padded +16B every 128B
// so the tile lands on the LDS_STRIDE(72-half) layout.  Wave-level op (EXEC
// ignored): call from one wave only; fence with s_wait_tensorcnt.
__device__ __forceinline__ void tdm_load_tile_f16(const _Float16* gsrc,
                                                  unsigned ldsOff,
                                                  unsigned rows, unsigned kHalves,
                                                  unsigned gstride) {
  const unsigned long long ga = (unsigned long long)(const void*)gsrc;
  v4u g0;
  g0.x = 1u;                                          // count=1 user descriptor
  g0.y = ldsOff;                                      // lds_addr
  g0.z = (unsigned)(ga & 0xFFFFFFFFu);                // global_addr[31:0]
  g0.w = (unsigned)((ga >> 32) & 0x01FFFFFFu) | (2u << 30);  // addr[56:32], type=2
  // group1: data_size=2B (code 1); pad_enable, pad every 32 DWORDs (code 4)
  // by 4 DWORDs (code 3)  => LDS row pitch 128B+16B = 144B = 72 halves.
  const unsigned w0 = 0x00010000u | (1u << 20) | (4u << 22) | (3u << 25);
  const unsigned td0 = gstride;          // tensor_dim0 (elements per row)
  const unsigned td1 = 0x10000u;         // tensor_dim1 (rows; generous, OOB-safe)
  v8i g1;
  g1[0] = (int)w0;
  g1[1] = (int)((td0 & 0xFFFFu) << 16);                       // dim0[15:0] @ bit48
  g1[2] = (int)(((td0 >> 16) & 0xFFFFu) | ((td1 & 0xFFFFu) << 16));
  g1[3] = (int)(((td1 >> 16) & 0xFFFFu) | ((kHalves & 0xFFFFu) << 16)); // tile_dim0
  g1[4] = (int)(rows & 0xFFFFu);                              // tile_dim1 (tile_dim2=0)
  g1[5] = (int)gstride;                                       // tensor_dim0_stride[31:0]
  g1[6] = 0;
  g1[7] = 0;
  const v4i z4 = {0, 0, 0, 0};
#if defined(__clang_major__) && (__clang_major__ >= 23)
  const v8i z8 = {0, 0, 0, 0, 0, 0, 0, 0};
  __builtin_amdgcn_tensor_load_to_lds(g0, g1, z4, z4, z8, 0);
#else
  __builtin_amdgcn_tensor_load_to_lds(g0, g1, z4, z4, 0);
#endif
}

// ============================ Kernel 1: gate_up + activation =================
// Block: 64 tokens x 64 h-cols (= 128 interleaved gate_up cols).  8 waves:
// 4 (M strips of 16) x 2 (64 gate_up cols -> 4 accumulators); K-step 64.
__launch_bounds__(256)
__global__ void k1_gate_up(const float* __restrict__ x,
                           const float* __restrict__ gup,
                           const float* __restrict__ gub,
                           const int*   __restrict__ counts,
                           _Float16*    __restrict__ hbuf) {
  __shared__ alignas(16) _Float16 Alds[64][LDS_STRIDE];    //  9.0 KB
  __shared__ alignas(16) _Float16 Blds[128][LDS_STRIDE];   // 18.0 KB [col][K]
  __shared__ alignas(16) float    Clds[64][132];           // 33.8 KB

  const int tid  = threadIdx.x;
  const int lane = tid & 31;
  const int wave = tid >> 5;
  const int wm   = wave & 3;
  const int wn   = wave >> 2;

  const int hcol0 = blockIdx.x * 64;
  const int row0  = blockIdx.y * 64;
  const int gcol0 = hcol0 * 2;

  const int e = expert_of_row(counts, row0);
  const float* W = gup + (size_t)e * HDIM * (2 * IDIM);

  v8f acc[4] = {};

  const int ar = tid >> 2;              // A row 0..63
  const int ak = (tid & 3) * 16;        // A K chunk of 16
  const int bp = tid & 31;              // B K-pair index (K = 2*bp, 2*bp+1)
  const int bc = (tid >> 5) * 16;       // B 16-col group

  for (int k0 = 0; k0 < HDIM; k0 += KSTEP) {
    // ---- stage A (packed b32 stores: 2 K halves per store) ----
    const float* asrc = x + (size_t)(row0 + ar) * HDIM + k0 + ak;
    __builtin_prefetch(asrc + KSTEP, 0, 1);
#pragma unroll
    for (int i = 0; i < 8; ++i)
      *reinterpret_cast<unsigned*>(&Alds[ar][ak + 2 * i]) =
          pack2(asrc[2 * i], asrc[2 * i + 1]);
    // ---- stage B transposed (pack K pair per column -> 1 b32 store) ----
    const float* b0 = W + (size_t)(k0 + 2 * bp) * (2 * IDIM) + gcol0 + bc;
    const float* b1 = b0 + 2 * IDIM;
    __builtin_prefetch(b0 + (size_t)KSTEP * (2 * IDIM), 0, 1);
#pragma unroll
    for (int i = 0; i < 16; ++i)
      *reinterpret_cast<unsigned*>(&Blds[bc + i][2 * bp]) = pack2(b0[i], b1[i]);
    __syncthreads();

    // ---- 8 WMMAs per stage ----
    const v16h a0 = load_frag(&Alds[wm * 16][0], lane);
    const v16h a1 = load_frag(&Alds[wm * 16][32], lane);
#pragma unroll
    for (int t = 0; t < 4; ++t) {
      const v16h bb0 = load_frag(&Blds[wn * 64 + t * 16][0], lane);
      acc[t] = wmma16(a0, bb0, acc[t]);
      const v16h bb1 = load_frag(&Blds[wn * 64 + t * 16][32], lane);
      acc[t] = wmma16(a1, bb1, acc[t]);
    }
    __syncthreads();
  }

  // Spill C to LDS (C layout: M = vgpr + 8*(lane>>4), N = lane&15).
  {
    const int r  = lane & 15;
    const int hi = lane >> 4;
#pragma unroll
    for (int t = 0; t < 4; ++t)
#pragma unroll
      for (int i = 0; i < 8; ++i)
        Clds[wm * 16 + hi * 8 + i][wn * 64 + t * 16 + r] = acc[t][i];
  }
  __syncthreads();

  // Bias + gated activation; pair even(gate)/odd(up) cols; store h as f16.
  {
    const int rr = tid >> 2;            // 0..63
    const int jj = (tid & 3) * 16;      // 16 h-cols per thread
    const float* bias = gub + (size_t)e * (2 * IDIM) + gcol0;
    _Float16* hdst = hbuf + (size_t)(row0 + rr) * IDIM + hcol0;
#pragma unroll
    for (int i = 0; i < 16; ++i) {
      const int j = jj + i;
      float g = Clds[rr][2 * j]     + bias[2 * j];
      float u = Clds[rr][2 * j + 1] + bias[2 * j + 1];
      g = fminf(g, LIMIT);
      u = fminf(fmaxf(u, -LIMIT), LIMIT);
      const float glu = g / (1.0f + __expf(-ALPHA * g));
      hdst[j] = (_Float16)((u + 1.0f) * glu);
    }
  }
}

// ============================ Kernel 2: down projection ======================
// Block: 64 tokens x 64 out cols.  8 waves: 4(M) x 2(N: 32 cols -> 2 accums).
// A tile (f16, no conversion needed) staged by the Tensor Data Mover.
__launch_bounds__(256)
__global__ void k2_down(const _Float16* __restrict__ hbuf,
                        const float* __restrict__ dw,
                        const float* __restrict__ db,
                        const int*   __restrict__ counts,
                        float*       __restrict__ out) {
  __shared__ alignas(16) _Float16 Alds[64][LDS_STRIDE];   // 9.0 KB (TDM dest)
  __shared__ alignas(16) _Float16 Blds[64][LDS_STRIDE];   // 9.0 KB [col][K]

  const int tid  = threadIdx.x;
  const int lane = tid & 31;
  const int wave = tid >> 5;
  const int wm   = wave & 3;
  const int wn   = wave >> 2;

  const int c0   = blockIdx.x * 64;
  const int row0 = blockIdx.y * 64;

  const int e = expert_of_row(counts, row0);
  const float* W = dw + (size_t)e * IDIM * HDIM;

  // LDS byte offset of Alds: shared aperture keeps the offset in the low 32b.
  const unsigned ldsOffA = (unsigned)(unsigned long long)(const void*)&Alds[0][0];

  v8f acc[2] = {};

  const int bp = tid & 31;              // B K-pair index
  const int bc = (tid >> 5) * 8;        // B 8-col group

  for (int k0 = 0; k0 < IDIM; k0 += KSTEP) {
    // ---- stage A via TDM (one DMA per stage, issued by wave 0) ----
    if (wave == 0)
      tdm_load_tile_f16(hbuf + (size_t)row0 * IDIM + k0, ldsOffA,
                        /*rows=*/64, /*kHalves=*/KSTEP, /*gstride=*/IDIM);
    // ---- stage B transposed with packed stores ----
    const float* b0 = W + (size_t)(k0 + 2 * bp) * HDIM + c0 + bc;
    const float* b1 = b0 + HDIM;
    __builtin_prefetch(b0 + (size_t)KSTEP * HDIM, 0, 1);
#pragma unroll
    for (int i = 0; i < 8; ++i)
      *reinterpret_cast<unsigned*>(&Blds[bc + i][2 * bp]) = pack2(b0[i], b1[i]);
    if (wave == 0) __builtin_amdgcn_s_wait_tensorcnt(0);
    __syncthreads();

    // ---- 4 WMMAs per stage ----
    const v16h a0 = load_frag(&Alds[wm * 16][0], lane);
    const v16h a1 = load_frag(&Alds[wm * 16][32], lane);
#pragma unroll
    for (int t = 0; t < 2; ++t) {
      const v16h bb0 = load_frag(&Blds[wn * 32 + t * 16][0], lane);
      acc[t] = wmma16(a0, bb0, acc[t]);
      const v16h bb1 = load_frag(&Blds[wn * 32 + t * 16][32], lane);
      acc[t] = wmma16(a1, bb1, acc[t]);
    }
    __syncthreads();
  }

  // Direct epilogue from accumulator layout + bias.
  const int r  = lane & 15;
  const int hi = lane >> 4;
  const float* bias = db + (size_t)e * HDIM;
#pragma unroll
  for (int t = 0; t < 2; ++t) {
    const int col = c0 + wn * 32 + t * 16 + r;
    const float bv = bias[col];
#pragma unroll
    for (int i = 0; i < 8; ++i) {
      const int row = row0 + wm * 16 + hi * 8 + i;
      out[(size_t)row * HDIM + col] = acc[t][i] + bv;
    }
  }
}

// ============================ Launch =========================================
extern "C" void kernel_launch(void* const* d_in, const int* in_sizes, int n_in,
                              void* d_out, int out_size, void* d_ws, size_t ws_size,
                              hipStream_t stream) {
  const float* x      = (const float*)d_in[0];
  const float* gup    = (const float*)d_in[1];
  const float* gub    = (const float*)d_in[2];
  const float* dw     = (const float*)d_in[3];
  const float* db     = (const float*)d_in[4];
  const int*   counts = (const int*)d_in[5];
  float*       out    = (float*)d_out;
  _Float16*    hbuf   = (_Float16*)d_ws;     // N x I f16 intermediate (~94 MB)

  const int ntok = in_sizes[0] / HDIM;       // N = 16384
  (void)n_in; (void)out_size; (void)ws_size;

  dim3 block(256);
  dim3 g1(IDIM / 64, ntok / 64);             // 45 x 256
  dim3 g2(HDIM / 64, ntok / 64);             // 45 x 256

  k1_gate_up<<<g1, block, 0, stream>>>(x, gup, gub, counts, hbuf);
  k2_down  <<<g2, block, 0, stream>>>(hbuf, dw, db, counts, out);
}
